// ScaledDotProductAttention_61692910240526
// MI455X (gfx1250) — matrix operand order repaired and checked by reading, not compiled, not run on hardware
//
#include <hip/hip_runtime.h>
#include <hip/hip_bf16.h>

// ---------------------------------------------------------------------------
// Problem constants
// ---------------------------------------------------------------------------
#define BB 4
#define SS 4096
#define EE 256
#define NC 16
#define ROWS (BB * SS)          // 16384 total query rows
#define LOG2E 1.44269504088896340736f
#define SCALING 0.0625f         // E^-0.5 = 1/16

typedef __bf16 v8bf  __attribute__((ext_vector_type(8)));
typedef __bf16 v16bf __attribute__((ext_vector_type(16)));
typedef float  v8f   __attribute__((ext_vector_type(8)));

union VecU { v16bf v; v8bf h[2]; };

static __device__ __forceinline__ v8f wmma_bf16(v16bf a, v16bf b, v8f c) {
    return __builtin_amdgcn_wmma_f32_16x16x32_bf16(
        /*neg_a=*/false, a, /*neg_b=*/false, b,
        /*c_mod=*/(short)0, c, /*reuse_a=*/false, /*reuse_b=*/false);
}

// A fragment (16x32 bf16, M x K): lane<16 -> row=lane, K chunks {0..7,16..23};
// lane>=16 -> row=lane-16, K chunks {8..15,24..31}.  p points at this lane's
// first 8-element chunk; second chunk is +16 elements.
static __device__ __forceinline__ v16bf ldA(const __bf16* p) {
    VecU u;
    u.h[0] = *(const v8bf*)(p);
    u.h[1] = *(const v8bf*)(p + 16);
    return u.v;
}

// B fragment (32x16 bf16, K x N): lane -> column N=lane%16, K chunks {0..7,16..23}
// (lane<16) / {8..15,24..31} (lane>=16); p at K row 8*(lane/16), ld = row stride.
static __device__ __forceinline__ v16bf ldB(const __bf16* p, int ld) {
    v16bf r;
#pragma unroll
    for (int j = 0; j < 8; ++j) {
        r[j] = p[j * ld];
        r[j + 8] = p[(16 + j) * ld];
    }
    return r;
}

// ---------------------------------------------------------------------------
// CDNA5 async global->LDS copy (ASYNCcnt path), with portable fallbacks.
// Builtin prototype (from toolchain diagnostic): pointee type is int4
// (vector_size(16)), src in address_space(1), dst in address_space(3).
// ---------------------------------------------------------------------------
typedef int v4i_async __attribute__((vector_size(16)));
typedef __attribute__((address_space(1))) v4i_async gv4i;
typedef __attribute__((address_space(3))) v4i_async lv4i;

static __device__ __forceinline__ void async_copy16(const void* g, void* l) {
#if __has_builtin(__builtin_amdgcn_global_load_async_to_lds_b128)
    __builtin_amdgcn_global_load_async_to_lds_b128((gv4i*)g, (lv4i*)l, 0, 0);
#else
    *(uint4*)l = *(const uint4*)g;   // synchronous fallback
#endif
}

static __device__ __forceinline__ void wait_async0() {
#if __has_builtin(__builtin_amdgcn_s_wait_asynccnt)
    __builtin_amdgcn_s_wait_asynccnt(0);
#else
#if __has_builtin(__builtin_amdgcn_global_load_async_to_lds_b128)
    asm volatile("s_wait_asynccnt 0" ::: "memory");
#endif
#endif
}

// ---------------------------------------------------------------------------
// Kernel 1: convert x (f32) -> bf16, 8 elements/thread
// ---------------------------------------------------------------------------
__global__ void k_convert_x(const float* __restrict__ x, __bf16* __restrict__ xb) {
    int i = (blockIdx.x * blockDim.x + threadIdx.x) * 8;
    v8f v = *(const v8f*)(x + i);
    *(v8bf*)(xb + i) = __builtin_convertvector(v, v8bf);
}

// ---------------------------------------------------------------------------
// Kernel 2: transpose + convert weight  w[O][E] (f32) -> wt[E][O] (bf16)
// ---------------------------------------------------------------------------
__global__ void k_transpose_w(const float* __restrict__ w, __bf16* __restrict__ wt,
                              int O, int E) {
    int idx = blockIdx.x * blockDim.x + threadIdx.x;
    if (idx >= O * E) return;
    int o = idx % O;
    int e = idx / O;
    wt[e * O + o] = (__bf16)w[o * E + e];
}

// ---------------------------------------------------------------------------
// Kernel 3: projection  Y[r][o] = sum_e xb[r][e]*wt[e][o] + bias[o]
// One wave = 16x16 output tile, K-loop of 8 (E=256).
// transK==0: write bf16 row-major [r][256]; transK==1: write K^T [b][e][s].
// ---------------------------------------------------------------------------
__global__ __launch_bounds__(128) void k_qkv_proj(
    const __bf16* __restrict__ xb, const __bf16* __restrict__ wt,
    const float* __restrict__ bias, __bf16* __restrict__ out, int transK) {
    const int gw   = blockIdx.x * 4 + (threadIdx.x >> 5);
    const int lane = threadIdx.x & 31;
    const int ln   = lane & 15;
    const int lh   = lane >> 4;
    const int r0   = (gw >> 4) * 16;
    const int col0 = (gw & 15) * 16;

    v8f acc = {0.f, 0.f, 0.f, 0.f, 0.f, 0.f, 0.f, 0.f};
#pragma unroll
    for (int kk = 0; kk < 8; ++kk) {
        v16bf a = ldA(xb + (size_t)(r0 + ln) * EE + kk * 32 + lh * 8);
        v16bf b = ldB(wt + (size_t)(kk * 32 + lh * 8) * EE + col0 + ln, EE);
        acc = wmma_bf16(a, b, acc);
    }
    const float bb = bias[col0 + ln];
    acc = acc + bb;

    if (!transK) {
#pragma unroll
        for (int g = 0; g < 8; ++g) {
            int r = r0 + g + lh * 8;
            out[(size_t)r * EE + col0 + ln] = (__bf16)acc[g];
        }
    } else {
#pragma unroll
        for (int g = 0; g < 8; ++g) {
            int r = r0 + g + lh * 8;
            int b2 = r >> 12;          // batch
            int s  = r & 4095;         // seq pos
            out[(size_t)b2 * (EE * SS) + (size_t)(col0 + ln) * SS + s] = (__bf16)acc[g];
        }
    }
}

// ---------------------------------------------------------------------------
// Kernel 4: flash attention with async double-buffered K/V staging in LDS.
// Block = 4 waves = 64 consecutive queries (one batch); each wave owns 16.
// q: bf16 [r][256]; kt: bf16 [b][e][s]; v: bf16 [r][256];
// ao out: bf16 [r][256] = (softmax(QK^T * sc) * sc) @ V
//
// Dynamic LDS layout (68 KB):
//   kbuf[2]: 32 keys x 256 e, e-major  [e][32]   -> 16 KB each
//   vbuf[2]: 32 keys x 256 e, key-major [k][256] -> 16 KB each
//   psh    : 4 waves x 16x32 P-transpose tiles   ->  4 KB
// ---------------------------------------------------------------------------
__global__ __launch_bounds__(128) void k_flash_attn(
    const __bf16* __restrict__ q, const __bf16* __restrict__ kt,
    const __bf16* __restrict__ v, __bf16* __restrict__ ao) {
    extern __shared__ __align__(16) char smem[];
    __bf16* kbuf[2];
    __bf16* vbuf[2];
    kbuf[0] = (__bf16*)smem;
    kbuf[1] = kbuf[0] + 8192;
    vbuf[0] = kbuf[1] + 8192;
    vbuf[1] = vbuf[0] + 8192;
    __bf16* psh = vbuf[1] + 8192;

    const int tid  = threadIdx.x;
    const int wid  = tid >> 5;
    const int lane = tid & 31;
    const int ln   = lane & 15;
    const int lh   = lane >> 4;
    const int gw   = blockIdx.x * 4 + wid;
    const int r0   = gw * 16;           // first query row (global)
    const int b    = r0 >> 12;          // batch index
    const __bf16* ktb   = kt + (size_t)b * (EE * SS);
    const __bf16* vbase = v + (size_t)b * SS * EE;
    __bf16* pl = psh + wid * (16 * 32);

    // Q fragments for the whole E=256 reduction, resident in registers.
    v16bf qa[8];
#pragma unroll
    for (int kk = 0; kk < 8; ++kk)
        qa[kk] = ldA(q + (size_t)(r0 + ln) * EE + kk * 32 + lh * 8);

    float m_i[8], l_i[8];
    v8f o_acc[16];
#pragma unroll
    for (int g = 0; g < 8; ++g) { m_i[g] = -__builtin_inff(); l_i[g] = 0.f; }
#pragma unroll
    for (int et = 0; et < 16; ++et)
        o_acc[et] = (v8f){0.f, 0.f, 0.f, 0.f, 0.f, 0.f, 0.f, 0.f};

    // Cooperative async fill of one 32-key chunk: K (strided rows of K^T) and
    // V (one contiguous 16 KB block).  8+8 b128 segments per thread.
    auto fill = [&](int buf, int key0) {
        __bf16* kd = kbuf[buf];
        __bf16* vd = vbuf[buf];
#pragma unroll
        for (int i = 0; i < 8; ++i) {
            int s = tid + 128 * i;             // 0..1023
            int e = s >> 2, part = s & 3;      // K: 256 rows x 4 16B segs
            async_copy16(ktb + (size_t)e * SS + key0 + part * 8,
                         kd + e * 32 + part * 8);
        }
        const __bf16* vg = vbase + (size_t)key0 * EE;
#pragma unroll
        for (int i = 0; i < 8; ++i) {
            int s = tid + 128 * i;             // contiguous 16 KB
            async_copy16(vg + s * 8, vd + s * 8);
        }
    };

    fill(0, 0);

    for (int c = 0; c < SS / 32; ++c) {
        const int cur = c & 1;
        const __bf16* kl = kbuf[cur];
        const __bf16* vl = vbuf[cur];

        wait_async0();
        __syncthreads();
        if (c + 1 < SS / 32) fill(cur ^ 1, (c + 1) * 32);

        // ---- scores: two 16x16 D tiles covering 32 keys (from LDS) ----
        v8f s0 = {0.f, 0.f, 0.f, 0.f, 0.f, 0.f, 0.f, 0.f};
        v8f s1 = {0.f, 0.f, 0.f, 0.f, 0.f, 0.f, 0.f, 0.f};
#pragma unroll
        for (int kk = 0; kk < 8; ++kk) {
            const __bf16* bp = kl + (kk * 32 + lh * 8) * 32 + ln;
            s0 = wmma_bf16(qa[kk], ldB(bp, 32),      s0);
            s1 = wmma_bf16(qa[kk], ldB(bp + 16, 32), s1);
        }
        s0 = s0 * SCALING;
        s1 = s1 * SCALING;

        // ---- online softmax (rows live across 16-lane halves) ----
        v8f p0, p1, corrv;
#pragma unroll
        for (int g = 0; g < 8; ++g) {
            float mx = fmaxf(s0[g], s1[g]);
            mx = fmaxf(mx, __shfl_xor(mx, 1));
            mx = fmaxf(mx, __shfl_xor(mx, 2));
            mx = fmaxf(mx, __shfl_xor(mx, 4));
            mx = fmaxf(mx, __shfl_xor(mx, 8));
            float mn = fmaxf(m_i[g], mx);
            float cr = exp2f((m_i[g] - mn) * LOG2E);
            float a0 = exp2f((s0[g] - mn) * LOG2E);
            float a1 = exp2f((s1[g] - mn) * LOG2E);
            float rs = a0 + a1;
            rs += __shfl_xor(rs, 1);
            rs += __shfl_xor(rs, 2);
            rs += __shfl_xor(rs, 4);
            rs += __shfl_xor(rs, 8);
            l_i[g] = l_i[g] * cr + rs;
            m_i[g] = mn;
            corrv[g] = cr;
            p0[g] = a0;
            p1[g] = a1;
        }
#pragma unroll
        for (int et = 0; et < 16; ++et)
            o_acc[et] = o_acc[et] * corrv;

        // ---- P: D-layout -> A-layout via per-wave LDS tile ----
#pragma unroll
        for (int g = 0; g < 8; ++g) {
            int row = g + lh * 8;
            pl[row * 32 + ln]      = (__bf16)p0[g];
            pl[row * 32 + 16 + ln] = (__bf16)p1[g];
        }
        v16bf pa = ldA(pl + ln * 32 + lh * 8);

        // ---- O += P @ V  (V fragments from LDS) ----
        const __bf16* vb = vl + lh * 8 * EE + ln;
#pragma unroll
        for (int et = 0; et < 16; ++et)
            o_acc[et] = wmma_bf16(pa, ldB(vb + et * 16, EE), o_acc[et]);

        __syncthreads();   // all waves done reading this buffer
    }

    // ---- epilogue: attn_out = O * scaling / l ----
    v8f scl;
#pragma unroll
    for (int g = 0; g < 8; ++g) scl[g] = SCALING / l_i[g];
#pragma unroll
    for (int et = 0; et < 16; ++et) {
        v8f ov = o_acc[et] * scl;
#pragma unroll
        for (int g = 0; g < 8; ++g) {
            int r = r0 + g + lh * 8;
            ao[(size_t)r * EE + et * 16 + ln] = (__bf16)ov[g];
        }
    }
}

// ---------------------------------------------------------------------------
// Kernel 5: output head  out[r][o] = sum_e ao[r][e]*wot[e][o] + bo[o]
// N = 16 = NUM_CLASSES exactly one WMMA tile; one wave per 16 rows.
// ---------------------------------------------------------------------------
__global__ __launch_bounds__(128) void k_out_proj(
    const __bf16* __restrict__ ao, const __bf16* __restrict__ wot,
    const float* __restrict__ bo, float* __restrict__ out) {
    const int gw   = blockIdx.x * 4 + (threadIdx.x >> 5);
    const int lane = threadIdx.x & 31;
    const int ln   = lane & 15;
    const int lh   = lane >> 4;
    const int r0   = gw * 16;

    v8f acc = {0.f, 0.f, 0.f, 0.f, 0.f, 0.f, 0.f, 0.f};
#pragma unroll
    for (int kk = 0; kk < 8; ++kk) {
        v16bf a = ldA(ao + (size_t)(r0 + ln) * EE + kk * 32 + lh * 8);
        v16bf b = ldB(wot + (size_t)(kk * 32 + lh * 8) * NC + ln, NC);
        acc = wmma_bf16(a, b, acc);
    }
    const float bb = bo[ln];
    acc = acc + bb;
#pragma unroll
    for (int g = 0; g < 8; ++g) {
        int r = r0 + g + lh * 8;
        out[(size_t)r * NC + ln] = acc[g];
    }
}

// ---------------------------------------------------------------------------
// Host launch
// ---------------------------------------------------------------------------
extern "C" void kernel_launch(void* const* d_in, const int* in_sizes, int n_in,
                              void* d_out, int out_size, void* d_ws, size_t ws_size,
                              hipStream_t stream) {
    const float* x  = (const float*)d_in[0];
    const float* wq = (const float*)d_in[1];
    const float* bq = (const float*)d_in[2];
    const float* wk = (const float*)d_in[3];
    const float* bk = (const float*)d_in[4];
    const float* wv = (const float*)d_in[5];
    const float* bv = (const float*)d_in[6];
    const float* wo = (const float*)d_in[7];
    const float* bo = (const float*)d_in[8];
    float* out = (float*)d_out;

    char* ws = (char*)d_ws;
    const size_t MB = 1024u * 1024u;
    __bf16* xb  = (__bf16*)(ws + 0 * MB);    // 8 MB  x in bf16
    __bf16* qw  = (__bf16*)(ws + 8 * MB);    // 8 MB  Q  [r][e]
    __bf16* ktw = (__bf16*)(ws + 16 * MB);   // 8 MB  K^T [b][e][s]
    __bf16* vw  = (__bf16*)(ws + 24 * MB);   // 8 MB  V  [r][e]
    __bf16* aow = (__bf16*)(ws + 32 * MB);   // 8 MB  attn_out [r][e]
    __bf16* wqt = (__bf16*)(ws + 40 * MB);              // 128 KB
    __bf16* wkt = (__bf16*)(ws + 40 * MB + 131072);     // 128 KB
    __bf16* wvt = (__bf16*)(ws + 40 * MB + 2 * 131072); // 128 KB
    __bf16* wot = (__bf16*)(ws + 40 * MB + 3 * 131072); // 8 KB

    // 1) precision / layout prep
    k_convert_x<<<(ROWS * EE / 8) / 256, 256, 0, stream>>>(x, xb);
    k_transpose_w<<<(EE * EE + 255) / 256, 256, 0, stream>>>(wq, wqt, EE, EE);
    k_transpose_w<<<(EE * EE + 255) / 256, 256, 0, stream>>>(wk, wkt, EE, EE);
    k_transpose_w<<<(EE * EE + 255) / 256, 256, 0, stream>>>(wv, wvt, EE, EE);
    k_transpose_w<<<(NC * EE + 255) / 256, 256, 0, stream>>>(wo, wot, NC, EE);

    // 2) QKV projections: 16384 waves each (1024 row-tiles x 16 col-tiles)
    k_qkv_proj<<<4096, 128, 0, stream>>>(xb, wqt, bq, qw, 0);
    k_qkv_proj<<<4096, 128, 0, stream>>>(xb, wkt, bk, ktw, 1);
    k_qkv_proj<<<4096, 128, 0, stream>>>(xb, wvt, bv, vw, 0);

    // 3) flash attention: 256 blocks x 4 waves, 68 KB dynamic LDS each
    k_flash_attn<<<256, 128, 68 * 1024, stream>>>(qw, ktw, vw, aow);

    // 4) output head: 1024 waves
    k_out_proj<<<256, 128, 68 * 0, stream>>>(aow, wot, bo, out);
}
